// TriangleMultiplication_12094627906017
// MI455X (gfx1250) — matrix operand order repaired and checked by reading, not compiled, not run on hardware
//
#include <hip/hip_runtime.h>

typedef __attribute__((ext_vector_type(16))) _Float16 v16h;
typedef __attribute__((ext_vector_type(8)))  _Float16 v8h;
typedef __attribute__((ext_vector_type(8)))  float    v8f;

#define NSP   512
#define NPIX  (512*512)          // 262144 pixels
#define CDIM  128
#define LNS   152                // halves: LDS row stride for 128-wide f16 tiles (16B aligned, bank-spread)
#define TRS   40                 // halves: LDS row stride for 32-wide f16 tiles in k_tri

// ---- workspace layout (bytes) ----
#define WS_AT   0ull                                        // f16 [128][512][512]  a[i,k,d] -> aT[d][i][k]
#define WS_BT   (WS_AT   + 128ull*NPIX*2ull)                // f16 [128][512][512]
#define WS_GATE (WS_BT   + 128ull*NPIX*2ull)                // f32 [NPIX][128]
#define WS_PROD (WS_GATE + (unsigned long long)NPIX*CDIM*4ull) // f32 [128][NPIX]
#define WS_W1T  (WS_PROD + (unsigned long long)NPIX*CDIM*4ull) // f16 [256][128]
#define WS_W2T  (WS_W1T  + 256ull*128ull*2ull)
#define WS_WGT  (WS_W2T  + 256ull*128ull*2ull)
#define WS_WOT  (WS_WGT  + 128ull*128ull*2ull)

#if __has_builtin(__builtin_amdgcn_tensor_load_to_lds) && __has_builtin(__builtin_amdgcn_s_wait_tensorcnt)
#define HAVE_TDM 1
#endif

static __device__ __forceinline__ v8f wmma_f16(v16h a, v16h b, v8f c) {
    return __builtin_amdgcn_wmma_f32_16x16x32_f16(false, a, false, b, (short)0, c, false, false);
}

// B fragment: lane n=l&15 holds column n, K = kc+8*(l>>4) .. +7 then kc+16+8*(l>>4) .. +7
static __device__ __forceinline__ v16h frag16(const _Float16* p) {
    v8h lo = *(const v8h*)p;
    v8h hi = *(const v8h*)(p + 16);
    return __builtin_shufflevector(lo, hi, 0,1,2,3,4,5,6,7,8,9,10,11,12,13,14,15);
}

// A fragment from row-major [row][k] tile (stride halves): lane m=l&15 row,
// halves [kc+8*ks .. +7] then [kc+16+8*ks .. +7]  (16-bit A 16x32 layout)
static __device__ __forceinline__ v16h fragA(const _Float16* base, int stride, int rowbase, int lane, int kc) {
    int m  = lane & 15;
    int ks = lane >> 4;
    const _Float16* p = base + (long)(rowbase + m) * stride + kc + 8 * ks;
    v8h lo = *(const v8h*)p;
    v8h hi = *(const v8h*)(p + 16);
    return __builtin_shufflevector(lo, hi, 0,1,2,3,4,5,6,7,8,9,10,11,12,13,14,15);
}

static __device__ __forceinline__ v16h fragB(const _Float16* base, int stride, int colbase, int lane, int kc) {
    int n  = lane & 15;
    int ks = lane >> 4;
    return frag16(base + (long)(colbase + n) * stride + kc + 8 * ks);
}

static __device__ __forceinline__ float sigmoidf(float x) {
    return 1.0f / (1.0f + __expf(-x));
}

#ifdef HAVE_TDM
typedef __attribute__((ext_vector_type(4))) unsigned int u32x4;
typedef __attribute__((ext_vector_type(8))) int i32x8;
typedef __attribute__((ext_vector_type(4))) int i32x4;

// TDM descriptor: 2-D tile, data_size=2B, tile 32(el) x 128(rows), tensor row
// stride 512 el, LDS padding of 4 DWORDs after every 16 DWORDs (-> 40-half rows).
// Bit layout per CDNA5 ISA 8.3/8.4.
static __device__ __forceinline__ void tdm_load_tile(unsigned lds_byte, const _Float16* gsrc)
{
    unsigned long long ga = (unsigned long long)(const void*)gsrc;
    u32x4 g0;
    g0[0] = 1u;                                             // count=1 (valid user D#)
    g0[1] = lds_byte;                                       // lds_addr
    g0[2] = (unsigned)(ga & 0xFFFFFFFFull);                 // global_addr[31:0]
    g0[3] = (unsigned)((ga >> 32) & 0x01FFFFFFull) | (2u << 30); // global_addr[56:32] | type=2
    i32x8 g1;
    g1[0] = (int)((1u << 16) | (1u << 20) | (3u << 22) | (3u << 25)); // data_size=2B | pad_en | interval=16DW | amount=4DW
    g1[1] = (int)(512u << 16);        // tensor_dim0 = 512 (bits 79:48, low16 here)
    g1[2] = (int)(512u << 16);        // tensor_dim0 hi = 0 ; tensor_dim1 = 512 (low16)
    g1[3] = (int)(32u  << 16);        // tensor_dim1 hi = 0 ; tile_dim0 = 32
    g1[4] = 128;                      // tile_dim1 = 128 ; tile_dim2 = 0
    g1[5] = 512;                      // tensor_dim0_stride = 512 (low32)
    g1[6] = 0;                        // stride0 hi16 = 0 ; tensor_dim1_stride low16 = 0
    g1[7] = 4;                        // tensor_dim1_stride hi (262144 >> 16)
    i32x4 z4 = {0, 0, 0, 0};
    i32x8 z8 = {0, 0, 0, 0, 0, 0, 0, 0};
    __builtin_amdgcn_tensor_load_to_lds(g0, g1, z4, z4, z8, 0);
}
#endif

// ---------------- prep: transpose weights to f16 [col][k] ----------------
__global__ __launch_bounds__(256) void k_prep(
    const float* __restrict__ w1, const float* __restrict__ w2,
    const float* __restrict__ wg, const float* __restrict__ wo,
    _Float16* w1T, _Float16* w2T, _Float16* wgT, _Float16* woT)
{
    int idx = blockIdx.x * 256 + threadIdx.x;
    if (idx < 256 * 128) {
        int col = idx >> 7, k = idx & 127;
        w1T[idx] = (_Float16)w1[k * 256 + col];
    } else if (idx < 2 * 256 * 128) {
        int j = idx - 256 * 128; int col = j >> 7, k = j & 127;
        w2T[j] = (_Float16)w2[k * 256 + col];
    } else if (idx < 2 * 256 * 128 + 128 * 128) {
        int j = idx - 2 * 256 * 128; int col = j >> 7, k = j & 127;
        wgT[j] = (_Float16)wg[k * 128 + col];
    } else if (idx < 2 * 256 * 128 + 2 * 128 * 128) {
        int j = idx - (2 * 256 * 128 + 128 * 128); int col = j >> 7, k = j & 127;
        woT[j] = (_Float16)wo[k * 128 + col];
    }
}

// ---------------- stage 1: LN + projections + gating ----------------
__global__ __launch_bounds__(256) void k_proj(
    const float* __restrict__ z, const float* __restrict__ ng, const float* __restrict__ nb,
    const _Float16* __restrict__ w1T, const float* __restrict__ b1,
    const _Float16* __restrict__ w2T, const float* __restrict__ b2,
    const _Float16* __restrict__ wgT, const float* __restrict__ bg,
    _Float16* __restrict__ aT, _Float16* __restrict__ bT, float* __restrict__ gate)
{
    __shared__ _Float16 znh[16 * LNS];
    const int t    = threadIdx.x;
    const int lane = t & 31;
    const int w    = t >> 5;
    const int row  = t >> 4;
    const int g    = t & 15;
    const long pix0 = (long)blockIdx.x * 16;

    float v[8];
    const float* zp = z + (pix0 + row) * CDIM + g * 8;
    float s = 0.f, q = 0.f;
#pragma unroll
    for (int u = 0; u < 8; ++u) { float x = zp[u]; v[u] = x; s += x; q += x * x; }
#pragma unroll
    for (int m = 1; m < 16; m <<= 1) { s += __shfl_xor(s, m, 32); q += __shfl_xor(q, m, 32); }
    float mean = s * (1.0f / 128.0f);
    float var  = q * (1.0f / 128.0f) - mean * mean;
    float rstd = rsqrtf(var + 1e-5f);
#pragma unroll
    for (int u = 0; u < 8; ++u) {
        int c = g * 8 + u;
        znh[row * LNS + c] = (_Float16)((v[u] - mean) * rstd * ng[c] + nb[c]);
    }
    __syncthreads();

    v16h A[4];
#pragma unroll
    for (int kk = 0; kk < 4; ++kk) A[kk] = fragA(znh, LNS, 0, lane, 32 * kk);

    const int n  = lane & 15;
    const int hi = lane >> 4;

#pragma unroll
    for (int tq = 0; tq < 2; ++tq) {
        const int tt = 2 * w + tq;
        v8f acc1 = {0,0,0,0,0,0,0,0};
        v8f acc2 = {0,0,0,0,0,0,0,0};
#pragma unroll
        for (int kk = 0; kk < 4; ++kk) {
            v16h B1 = fragB(w1T, 128, tt * 16, lane, 32 * kk);
            acc1 = wmma_f16(A[kk], B1, acc1);
            v16h B2 = fragB(w2T, 128, tt * 16, lane, 32 * kk);
            acc2 = wmma_f16(A[kk], B2, acc2);
        }
        const float bb1 = b1[tt * 16 + n];
        const float bb2 = b2[tt * 16 + n];
        v8h oh;
#pragma unroll
        for (int r = 0; r < 8; ++r) {
            float p1v = acc1[r] + bb1;
            float p2v = sigmoidf(acc2[r] + bb2);
            oh[r] = (_Float16)(p2v * p1v);
        }
        const int d = (tt & 7) * 16 + n;
        _Float16* dst = (tt < 8 ? aT : bT) + (long)d * NPIX + pix0 + 8 * hi;
        *(v8h*)dst = oh;
    }

    {
        v8f acc = {0,0,0,0,0,0,0,0};
#pragma unroll
        for (int kk = 0; kk < 4; ++kk) {
            v16h B = fragB(wgT, 128, w * 16, lane, 32 * kk);
            acc = wmma_f16(A[kk], B, acc);
        }
        const int   col = w * 16 + n;
        const float bb  = bg[col];
#pragma unroll
        for (int r = 0; r < 8; ++r) {
            long pix = pix0 + r + 8 * hi;
            gate[pix * CDIM + col] = sigmoidf(acc[r] + bb);
        }
    }
}

// ---------------- stage 2: triangle einsum, per-channel GEMM ----------------
// grid (128 d, 4 iblk, 4 jblk); block 256 thr; C tile 128x128, K loop 512.
// LDS tiles staged by the Tensor Data Mover (double buffered) when available.
__global__ __launch_bounds__(256) void k_tri(
    const _Float16* __restrict__ aT, const _Float16* __restrict__ bT,
    float* __restrict__ prod)
{
    __shared__ _Float16 aL[2][128 * TRS];
    __shared__ _Float16 bL[2][128 * TRS];
    const int t = threadIdx.x, lane = t & 31, w = t >> 5;
    const int d  = blockIdx.x;
    const int i0 = blockIdx.y * 128;
    const int j0 = blockIdx.z * 128;
    const _Float16* ap = aT + (long)d * NPIX + (long)i0 * NSP;
    const _Float16* bp = bT + (long)d * NPIX + (long)j0 * NSP;

    v8f acc[8];
#pragma unroll
    for (int j = 0; j < 8; ++j) acc[j] = (v8f){0,0,0,0,0,0,0,0};

#ifdef HAVE_TDM
    // --- TDM double-buffered pipeline ---
    const unsigned aOff0 = (unsigned)(unsigned long long)(void*)&aL[0][0];
    const unsigned aOff1 = (unsigned)(unsigned long long)(void*)&aL[1][0];
    const unsigned bOff0 = (unsigned)(unsigned long long)(void*)&bL[0][0];
    const unsigned bOff1 = (unsigned)(unsigned long long)(void*)&bL[1][0];

    if (w == 0)      tdm_load_tile(aOff0, ap);
    else if (w == 1) tdm_load_tile(bOff0, bp);

    for (int c = 0; c < 16; ++c) {
        const int cur = c & 1;
        if (c + 1 < 16) {
            if (w == 0)      tdm_load_tile(cur ? aOff0 : aOff1, ap + (c + 1) * 32);
            else if (w == 1) tdm_load_tile(cur ? bOff0 : bOff1, bp + (c + 1) * 32);
            if (w < 2) __builtin_amdgcn_s_wait_tensorcnt(1);   // chunk c landed, c+1 in flight
        } else if (w < 2) {
            __builtin_amdgcn_s_wait_tensorcnt(0);
        }
        __syncthreads();                                       // publish chunk c to all waves
        v16h A = fragA(&aL[cur][0], TRS, w * 16, lane, 0);
#pragma unroll
        for (int jt = 0; jt < 8; ++jt) {
            v16h B = fragB(&bL[cur][0], TRS, jt * 16, lane, 0);
            acc[jt] = wmma_f16(A, B, acc[jt]);
        }
        __syncthreads();                                       // done reading buf before overwrite
    }
#else
    // --- fallback: cooperative synchronous staging ---
    const int lrow = t >> 1;
    const int lsel = (t & 1) * 16;
    for (int k0 = 0; k0 < NSP; k0 += 32) {
        __syncthreads();
        *(v8h*)(&aL[0][0] + lrow * TRS + lsel) = *(const v8h*)(ap + (long)lrow * NSP + k0 + lsel);
        *(v8h*)(&bL[0][0] + lrow * TRS + lsel) = *(const v8h*)(bp + (long)lrow * NSP + k0 + lsel);
        if (k0 + 32 < NSP) {
            __builtin_prefetch(ap + (long)lrow * NSP + k0 + 32 + lsel, 0, 1);
            __builtin_prefetch(bp + (long)lrow * NSP + k0 + 32 + lsel, 0, 1);
        }
        __syncthreads();
        v16h A = fragA(&aL[0][0], TRS, w * 16, lane, 0);
#pragma unroll
        for (int jt = 0; jt < 8; ++jt) {
            v16h B = fragB(&bL[0][0], TRS, jt * 16, lane, 0);
            acc[jt] = wmma_f16(A, B, acc[jt]);
        }
    }
#endif

    const int m  = lane & 15;
    const int hi = lane >> 4;
    const int ii = i0 + w * 16;
    float* pd = prod + (long)d * NPIX;
#pragma unroll
    for (int jt = 0; jt < 8; ++jt) {
        const int j = j0 + jt * 16 + m;
#pragma unroll
        for (int r = 0; r < 8; ++r) {
            pd[(long)(ii + r + 8 * hi) * NSP + j] = acc[jt][r];
        }
    }
}

// ---------------- stage 3: out-LN + wo projection + gate ----------------
__global__ __launch_bounds__(256) void k_out(
    const float* __restrict__ prod, const float* __restrict__ og, const float* __restrict__ ob,
    const _Float16* __restrict__ woT, const float* __restrict__ bo,
    const float* __restrict__ gate, float* __restrict__ out)
{
    __shared__ _Float16 pth[16 * LNS];
    __shared__ float redS[256];
    __shared__ float redQ[256];
    __shared__ float mstd[16][2];
    const int t = threadIdx.x, lane = t & 31, w = t >> 5;
    const long pix0 = (long)blockIdx.x * 16;
    const int row = t & 15;
    const int cg  = t >> 4;

    float v[8]; float s = 0.f, q = 0.f;
#pragma unroll
    for (int u = 0; u < 8; ++u) {
        float x = prod[(long)(cg * 8 + u) * NPIX + pix0 + row];
        v[u] = x; s += x; q += x * x;
    }
    redS[t] = s; redQ[t] = q;
    __syncthreads();
    if (t < 16) {
        float S = 0.f, Q = 0.f;
#pragma unroll
        for (int j = 0; j < 16; ++j) { S += redS[t + 16 * j]; Q += redQ[t + 16 * j]; }
        float mean = S * (1.0f / 128.0f);
        float var  = Q * (1.0f / 128.0f) - mean * mean;
        mstd[t][0] = mean;
        mstd[t][1] = rsqrtf(var + 1e-5f);
    }
    __syncthreads();
    const float mean = mstd[row][0], rstd = mstd[row][1];
#pragma unroll
    for (int u = 0; u < 8; ++u) {
        int c = cg * 8 + u;
        pth[row * LNS + c] = (_Float16)((v[u] - mean) * rstd * og[c] + ob[c]);
    }
    __syncthreads();

    v16h A[4];
#pragma unroll
    for (int kk = 0; kk < 4; ++kk) A[kk] = fragA(pth, LNS, 0, lane, 32 * kk);

    const int n = lane & 15, hi = lane >> 4;
    v8f acc = {0,0,0,0,0,0,0,0};
#pragma unroll
    for (int kk = 0; kk < 4; ++kk) {
        v16h B = fragB(woT, 128, w * 16, lane, 32 * kk);
        acc = wmma_f16(A[kk], B, acc);
    }
    const int   col = w * 16 + n;
    const float bb  = bo[col];
#pragma unroll
    for (int r = 0; r < 8; ++r) {
        long pix = pix0 + r + 8 * hi;
        float gv = gate[pix * CDIM + col];
        __builtin_nontemporal_store(gv * (acc[r] + bb), &out[pix * CDIM + col]);
    }
}

extern "C" void kernel_launch(void* const* d_in, const int* in_sizes, int n_in,
                              void* d_out, int out_size, void* d_ws, size_t ws_size,
                              hipStream_t stream)
{
    (void)in_sizes; (void)n_in; (void)out_size; (void)ws_size;
    const float* z  = (const float*)d_in[0];
    const float* ng = (const float*)d_in[1];
    const float* nb = (const float*)d_in[2];
    const float* w1 = (const float*)d_in[3];
    const float* b1 = (const float*)d_in[4];
    const float* w2 = (const float*)d_in[5];
    const float* b2 = (const float*)d_in[6];
    const float* wg = (const float*)d_in[7];
    const float* bg = (const float*)d_in[8];
    const float* og = (const float*)d_in[9];
    const float* ob = (const float*)d_in[10];
    const float* wo = (const float*)d_in[11];
    const float* bo = (const float*)d_in[12];

    char* ws = (char*)d_ws;
    _Float16* aT   = (_Float16*)(ws + WS_AT);
    _Float16* bT   = (_Float16*)(ws + WS_BT);
    float*    gate = (float*)   (ws + WS_GATE);
    float*    prod = (float*)   (ws + WS_PROD);
    _Float16* w1T  = (_Float16*)(ws + WS_W1T);
    _Float16* w2T  = (_Float16*)(ws + WS_W2T);
    _Float16* wgT  = (_Float16*)(ws + WS_WGT);
    _Float16* woT  = (_Float16*)(ws + WS_WOT);
    float* out = (float*)d_out;

    k_prep<<<384, 256, 0, stream>>>(w1, w2, wg, wo, w1T, w2T, wgT, woT);
    k_proj<<<NPIX / 16, 256, 0, stream>>>(z, ng, nb, w1T, b1, w2T, b2, wgT, bg, aT, bT, gate);
    dim3 g2(128, 4, 4);
    k_tri<<<g2, 256, 0, stream>>>(aT, bT, prod);
    k_out<<<NPIX / 16, 256, 0, stream>>>(prod, og, ob, woT, bo, gate, out);
}
